// CorrelationLayer_9457517986040
// MI455X (gfx1250) — matrix operand order repaired and checked by reading, not compiled, not run on hardware
//
#include <hip/hip_runtime.h>

#define BATCH 4
#define CH    256
#define HW    4096
#define BM    128
#define BN    128
#define BK    32
#define LDSR  40   // LDS row stride in bf16 elements: 80B rows -> b64-store alignment,
                   // 20-bank stride (conflict-free within 16-lane fragment groups)

typedef __attribute__((ext_vector_type(16))) __bf16 v16bf;
typedef __attribute__((ext_vector_type(8)))  float  v8f;

union FragU { v16bf v; unsigned u[8]; };

static __device__ __forceinline__ unsigned short f32_to_bf16(float f) {
    unsigned u = __builtin_bit_cast(unsigned, f);
    u += 0x7FFFu + ((u >> 16) & 1u);           // round-to-nearest-even
    return (unsigned short)(u >> 16);
}
static __device__ __forceinline__ float bf16_to_f32(unsigned short h) {
    unsigned u = ((unsigned)h) << 16;
    return __builtin_bit_cast(float, u);
}

// ---------------------------------------------------------------------------
// GEMM: corr[b,i,j] = sum_c f1[b,c,i] * f2[b,c,j], bf16x3 split precision.
// Grid: 4 batches * 32 * 32 tiles of 128x128. Block = 256 threads = 8 waves.
// Double-buffered LDS; per-thread 4(k)x4(m) staging blocks so the transposed
// hi/lo planes are written with ds_store_b64 (16/thread/K-step, not 64 b16).
// ---------------------------------------------------------------------------
__global__ __launch_bounds__(256) void corr_gemm(const float* __restrict__ f1,
                                                 const float* __restrict__ f2,
                                                 float* __restrict__ out) {
    // [buffer][hi/lo] planes, transposed [m][k] layout, row stride LDSR
    __shared__ __align__(16) unsigned short sA[2][2][BM * LDSR];
    __shared__ __align__(16) unsigned short sB[2][2][BN * LDSR];

    const int blk = blockIdx.x;
    const int b   = blk >> 10;            // 1024 tiles per batch
    const int rem = blk & 1023;
    const int tm  = (rem >> 5) << 7;      // i-tile base
    const int tn  = (rem & 31) << 7;      // j-tile base

    const float* gA = f1 + (size_t)b * CH * HW + tm;  // (k,m): gA[k*HW + m]
    const float* gB = f2 + (size_t)b * CH * HW + tn;  // (k,n): gB[k*HW + n]

    const int tid   = threadIdx.x;
    const int lane  = tid & 31;
    const int wave  = tid >> 5;
    const int m_off = (wave & 3) << 5;    // 0,32,64,96
    const int n_off = (wave >> 2) << 6;   // 0,64
    const int l15   = lane & 15;
    const int hi16  = (lane >> 4) & 1;
    const int a0    = hi16 ? 4 : 0;       // A K-half (u32 index) per ISA layout
    const int b0    = hi16 ? 4 : 0;       // B K-half (u32 index) per ISA layout

    // Staging assignment: wave owns k-rows kg..kg+3, lane owns cols c4..c4+3.
    const int kg = wave << 2;             // 0,4,...,28
    const int c4 = lane << 2;             // 0,4,...,124

    v8f acc[2][4];
    #pragma unroll
    for (int mi = 0; mi < 2; ++mi)
        #pragma unroll
        for (int ni = 0; ni < 4; ++ni) {
            v8f z = {};
            acc[mi][ni] = z;
        }

    // Prologue: global loads for K-step 0 (each wave: 4 contiguous 512B rows).
    float va[4][4], vb[4][4];
    #pragma unroll
    for (int q = 0; q < 4; ++q) {
        float4 ta = *(const float4*)(gA + (size_t)(kg + q) * HW + c4);
        float4 tb = *(const float4*)(gB + (size_t)(kg + q) * HW + c4);
        va[q][0] = ta.x; va[q][1] = ta.y; va[q][2] = ta.z; va[q][3] = ta.w;
        vb[q][0] = tb.x; vb[q][1] = tb.y; vb[q][2] = tb.z; vb[q][3] = tb.w;
    }

    for (int t = 0; t < CH / BK; ++t) {
        const int buf = t & 1;
        unsigned short* aH = sA[buf][0];
        unsigned short* aL = sA[buf][1];
        unsigned short* bH = sB[buf][0];
        unsigned short* bL = sB[buf][1];

        // Convert staging registers -> hi/lo bf16, transposed, b64 stores.
        #pragma unroll
        for (int e = 0; e < 4; ++e) {
            const int m = c4 + e;
            unsigned ahp[2], alp[2], bhp[2], blp[2];
            #pragma unroll
            for (int h = 0; h < 2; ++h) {
                unsigned short a0h = f32_to_bf16(va[2 * h][e]);
                unsigned short a1h = f32_to_bf16(va[2 * h + 1][e]);
                unsigned short a0l = f32_to_bf16(va[2 * h][e]     - bf16_to_f32(a0h));
                unsigned short a1l = f32_to_bf16(va[2 * h + 1][e] - bf16_to_f32(a1h));
                ahp[h] = (unsigned)a0h | ((unsigned)a1h << 16);
                alp[h] = (unsigned)a0l | ((unsigned)a1l << 16);
                unsigned short b0h = f32_to_bf16(vb[2 * h][e]);
                unsigned short b1h = f32_to_bf16(vb[2 * h + 1][e]);
                unsigned short b0l = f32_to_bf16(vb[2 * h][e]     - bf16_to_f32(b0h));
                unsigned short b1l = f32_to_bf16(vb[2 * h + 1][e] - bf16_to_f32(b1h));
                bhp[h] = (unsigned)b0h | ((unsigned)b1h << 16);
                blp[h] = (unsigned)b0l | ((unsigned)b1l << 16);
            }
            *(uint2*)&aH[m * LDSR + kg] = make_uint2(ahp[0], ahp[1]);
            *(uint2*)&aL[m * LDSR + kg] = make_uint2(alp[0], alp[1]);
            *(uint2*)&bH[m * LDSR + kg] = make_uint2(bhp[0], bhp[1]);
            *(uint2*)&bL[m * LDSR + kg] = make_uint2(blp[0], blp[1]);
        }
        __syncthreads();

        // Kick off global loads for the next K-step (hidden behind WMMAs).
        if (t < CH / BK - 1) {
            const int k0 = (t + 1) * BK;
            #pragma unroll
            for (int q = 0; q < 4; ++q) {
                float4 ta = *(const float4*)(gA + (size_t)(k0 + kg + q) * HW + c4);
                float4 tb = *(const float4*)(gB + (size_t)(k0 + kg + q) * HW + c4);
                va[q][0] = ta.x; va[q][1] = ta.y; va[q][2] = ta.z; va[q][3] = ta.w;
                vb[q][0] = tb.x; vb[q][1] = tb.y; vb[q][2] = tb.z; vb[q][3] = tb.w;
            }
        }

        // Preload ALL fragments for this K-step.
        // A (16x32 bf16 ISA layout): lanes<16 K=0-7,16-23; lanes>=16 K=8-15,24-31
        FragU ah[2], al[2], bh[4], bl[4];
        #pragma unroll
        for (int mi = 0; mi < 2; ++mi) {
            const unsigned* ph = (const unsigned*)&aH[(m_off + mi * 16 + l15) * LDSR];
            const unsigned* pl = (const unsigned*)&aL[(m_off + mi * 16 + l15) * LDSR];
            #pragma unroll
            for (int v = 0; v < 4; ++v) {
                ah[mi].u[v]     = ph[a0 + v];
                al[mi].u[v]     = pl[a0 + v];
                ah[mi].u[4 + v] = ph[a0 + 8 + v];
                al[mi].u[4 + v] = pl[a0 + 8 + v];
            }
        }
        // B (32x16 bf16 ISA layout): lanes<16 K=0-7,16-23; lanes>=16 K=8-15,24-31
        #pragma unroll
        for (int ni = 0; ni < 4; ++ni) {
            const unsigned* ph = (const unsigned*)&bH[(n_off + ni * 16 + l15) * LDSR];
            const unsigned* pl = (const unsigned*)&bL[(n_off + ni * 16 + l15) * LDSR];
            #pragma unroll
            for (int v = 0; v < 4; ++v) { bh[ni].u[v] = ph[b0 + v]; bl[ni].u[v] = pl[b0 + v]; bh[ni].u[4 + v] = ph[b0 + 8 + v]; bl[ni].u[4 + v] = pl[b0 + 8 + v]; }
        }

        // 24 WMMAs, product-outermost: same-accumulator distance = 8.
        #pragma unroll
        for (int ni = 0; ni < 4; ++ni)
            #pragma unroll
            for (int mi = 0; mi < 2; ++mi)
                acc[mi][ni] = __builtin_amdgcn_wmma_f32_16x16x32_bf16(
                    false, ah[mi].v, false, bh[ni].v, (short)0, acc[mi][ni], false, false);
        #pragma unroll
        for (int ni = 0; ni < 4; ++ni)
            #pragma unroll
            for (int mi = 0; mi < 2; ++mi)
                acc[mi][ni] = __builtin_amdgcn_wmma_f32_16x16x32_bf16(
                    false, ah[mi].v, false, bl[ni].v, (short)0, acc[mi][ni], false, false);
        #pragma unroll
        for (int ni = 0; ni < 4; ++ni)
            #pragma unroll
            for (int mi = 0; mi < 2; ++mi)
                acc[mi][ni] = __builtin_amdgcn_wmma_f32_16x16x32_bf16(
                    false, al[mi].v, false, bh[ni].v, (short)0, acc[mi][ni], false, false);
    }

    // Epilogue: C/D layout — VGPR r holds row (r | r+8), col = lane&15.
    float* outb = out + (size_t)b * HW * HW;
    const int row_add = hi16 ? 8 : 0;
    #pragma unroll
    for (int mi = 0; mi < 2; ++mi)
        #pragma unroll
        for (int ni = 0; ni < 4; ++ni) {
            int i0 = tm + m_off + mi * 16 + row_add;
            int j  = tn + n_off + ni * 16 + l15;
            #pragma unroll
            for (int r = 0; r < 8; ++r)
                outb[(size_t)(i0 + r) * HW + j] = acc[mi][ni][r];
        }
}

// ---------------------------------------------------------------------------
// In-place softmax over axis i for each column (b, j).
// Block = 256 threads: 32 columns per block (coalesced 128B per wave),
// 8 row-segments of 512 rows combined through LDS (online max/sum).
// Grid = BATCH*HW/32 = 512 blocks.
// ---------------------------------------------------------------------------
__global__ __launch_bounds__(256) void softmax_cols(float* __restrict__ out) {
    __shared__ float sm[8][32];
    __shared__ float ss[8][32];

    const int tid  = threadIdx.x;
    const int lane = tid & 31;
    const int seg  = tid >> 5;
    const int col  = blockIdx.x * 32 + lane;       // 0..BATCH*HW-1
    const int b    = col >> 12;
    const int j    = col & (HW - 1);

    float* base = out + (size_t)b * HW * HW + j;

    float m = -3.402823466e38f, s = 0.f;
    const int i0 = seg * 512;
    for (int i = i0; i < i0 + 512; ++i) {
        float x  = base[(size_t)i * HW];
        float nm = fmaxf(m, x);
        s = s * __expf(m - nm) + __expf(x - nm);
        m = nm;
    }
    sm[seg][lane] = m;
    ss[seg][lane] = s;
    __syncthreads();

    float M = -3.402823466e38f, S = 0.f;
    #pragma unroll
    for (int r = 0; r < 8; ++r) {
        float mr = sm[r][lane], sr = ss[r][lane];
        float nm = fmaxf(M, mr);
        S = S * __expf(M - nm) + sr * __expf(mr - nm);
        M = nm;
    }
    const float inv = 1.f / S;

    for (int i = i0; i < i0 + 512; ++i) {
        size_t off = (size_t)i * HW;
        base[off] = __expf(base[off] - M) * inv;
    }
}

extern "C" void kernel_launch(void* const* d_in, const int* in_sizes, int n_in,
                              void* d_out, int out_size, void* d_ws, size_t ws_size,
                              hipStream_t stream) {
    (void)in_sizes; (void)n_in; (void)out_size; (void)d_ws; (void)ws_size;
    const float* f1 = (const float*)d_in[0];
    const float* f2 = (const float*)d_in[1];
    float* out = (float*)d_out;

    corr_gemm<<<BATCH * (HW / BM) * (HW / BN), 256, 0, stream>>>(f1, f2, out);
    softmax_cols<<<(BATCH * HW) / 32, 256, 0, stream>>>(out);
}
